// MultiHeadAttention_91268055040488
// MI455X (gfx1250) — matrix operand order repaired and checked by reading, not compiled, not run on hardware
//
#include <hip/hip_runtime.h>

// Problem constants (match reference)
constexpr int Bc  = 2;
constexpr int Sc  = 2048;
constexpr int Hc  = 1024;
constexpr int NHc = 16;
constexpr int HDc = 64;
constexpr int KD  = 1024;   // GEMM contraction dim (== Hc)

typedef __attribute__((ext_vector_type(16))) _Float16 v16h;
typedef __attribute__((ext_vector_type(8)))  _Float16 v8h;
typedef __attribute__((ext_vector_type(4)))  _Float16 v4h;
typedef __attribute__((ext_vector_type(8)))  float    v8f;
typedef __attribute__((ext_vector_type(4)))  unsigned int v4u;
typedef __attribute__((ext_vector_type(8)))  int      v8i;
typedef __attribute__((ext_vector_type(4)))  int      v4i;

#define DEVI __device__ __forceinline__

DEVI v8f wmma_f16(v16h a, v16h b, v8f c) {
  // (neg_a, A, neg_b, B, c_mod, C, reuse_a, reuse_b)
  return __builtin_amdgcn_wmma_f32_16x16x32_f16(false, a, false, b, (short)0, c,
                                                false, false);
}

// A-fragment (16x32 f16, M x K), contiguous K order (matches B):
// lanes 0-15 : row m = lane,    K = {0..15} in halves 0..15
// lanes 16-31: row m = lane-16, K = {16..31}
DEVI v16h load_afrag(const _Float16* p, int kbA) {
  v8h lo = *(const v8h*)(p + kbA);
  v8h hi = *(const v8h*)(p + kbA + 8);
  v16h a;
#pragma unroll
  for (int i = 0; i < 8; ++i) { a[i] = lo[i]; a[i + 8] = hi[i]; }
  return a;
}

DEVI float redmax16(float v) {
#pragma unroll
  for (int m = 8; m >= 1; m >>= 1) v = fmaxf(v, __shfl_xor(v, m, 32));
  return v;
}
DEVI float redsum16(float v) {
#pragma unroll
  for (int m = 8; m >= 1; m >>= 1) v += __shfl_xor(v, m, 32);
  return v;
}

// LDS byte offset of a __shared__ pointer (generic addr low 32 bits = LDS offset)
DEVI unsigned lds_off(const void* p) { return (unsigned)(size_t)p; }

// ---------------------------------------------------------------------------
// Tensor Data Mover: async 2-D f16 tile load Global -> LDS.
// D# per CDNA5 ISA (§8.3 group0, §8.4 group1):
//   tile_d0 elements contiguous per row, tile_d1 rows, row stride = stride0
//   (all in 2-byte elements). LDS receives the tile packed row-major.
// ---------------------------------------------------------------------------
DEVI void tdm_load_2d(unsigned ldsaddr, const void* gptr,
                      unsigned tile_d0, unsigned tile_d1, unsigned stride0) {
  unsigned long long ga = (unsigned long long)(size_t)gptr;
  v4u g0;
  g0.x = 1u;                                    // count=1, user descriptor
  g0.y = ldsaddr;                               // lds_addr (bytes)
  g0.z = (unsigned)ga;                          // global_addr[31:0]
  g0.w = (unsigned)((ga >> 32) & 0x01FFFFFFu)   // global_addr[56:32]
         | (2u << 30);                          // type = 2 ("image")
  v8i g1;
  g1[0] = (int)(1u << 16);                      // data_size=1 (2B), no multicast
  g1[1] = (int)((stride0 & 0xFFFFu) << 16);     // tensor_dim0[15:0]
  g1[2] = (int)((stride0 >> 16) & 0xFFFFu);     // tensor_dim0[31:16] | dim1[15:0]=0
  g1[3] = (int)(0x10u | ((tile_d0 & 0xFFFFu) << 16)); // dim1[31:16]=0x10 (1M) | tile_dim0
  g1[4] = (int)(tile_d1 & 0xFFFFu);             // tile_dim1 (tile_dim2 = 0)
  g1[5] = (int)stride0;                         // tensor_dim0_stride[31:0]
  g1[6] = 0;                                    // stride hi / dim1_stride lo
  g1[7] = 0;
  v4i z = {};
#if __clang_major__ >= 23
  v8i z8 = {};
  __builtin_amdgcn_tensor_load_to_lds(g0, g1, z, z, z8, 0);
#else
  __builtin_amdgcn_tensor_load_to_lds(g0, g1, z, z, 0);
#endif
}

// ---------------------------------------------------------------------------
// fp32 -> f16 conversion (n multiple of 4)
// ---------------------------------------------------------------------------
__global__ __launch_bounds__(256)
void cvt_f32_f16(const float* __restrict__ s, _Float16* __restrict__ d, size_t n) {
  size_t i = ((size_t)blockIdx.x * 256 + threadIdx.x) * 4;
  if (i + 3 < n) {
    float4 f = *(const float4*)(s + i);
    v4h o;
    o[0] = (_Float16)f.x; o[1] = (_Float16)f.y;
    o[2] = (_Float16)f.z; o[3] = (_Float16)f.w;
    *(v4h*)(d + i) = o;
  }
}

// ---------------------------------------------------------------------------
// WMMA GEMM: Y(4096 x 1024) = X(4096 x 1024) * W^T + bias
// TDM double-buffers the 256x32 A tile and the 64x32 W tile in LDS; all 8
// waves of the block share the W tile (8x global-traffic reduction).
// MODE 0: f16 [b,h,s,d] (Q,K)  MODE 1: f16 [b,h,d,s] (V)  MODE 2: f32 [r,c]
// ---------------------------------------------------------------------------
template <int MODE>
__global__ __launch_bounds__(256)
void gemm_wmma(const _Float16* __restrict__ X, const _Float16* __restrict__ W,
               const float* __restrict__ bias, void* __restrict__ dst) {
  __shared__ __align__(128) _Float16 ldsA[2][256 * 32];  // 32 KB
  __shared__ __align__(128) _Float16 ldsW[2][64 * 32];   //  8 KB

  const int lane = threadIdx.x & 31;
  const int wid  = threadIdx.x >> 5;
  const int hseg = lane >> 4;
  const int l16  = lane & 15;
  const int kbA  = hseg ? 16 : 0;
  const int kbB  = hseg ? 16 : 0;

  const int rowB = blockIdx.x * 256;     // block row base
  const int row0 = rowB + wid * 32;      // this wave's rows
  const int col0 = blockIdx.y * 64;

  v8f c[2][4];
#pragma unroll
  for (int mi = 0; mi < 2; ++mi)
#pragma unroll
    for (int ni = 0; ni < 4; ++ni) c[mi][ni] = (v8f){};

  const _Float16* Xb = X + (size_t)rowB * KD;
  const _Float16* Wb = W + (size_t)col0 * KD;

  if (wid == 0) {  // prologue: fill buffer 0
    tdm_load_2d(lds_off(&ldsA[0][0]), Xb, 32, 256, KD);
    tdm_load_2d(lds_off(&ldsW[0][0]), Wb, 32, 64, KD);
  }

  for (int kb = 0; kb < KD; kb += 32) {
    const int buf = (kb >> 5) & 1;
    if (wid == 0) __builtin_amdgcn_s_wait_tensorcnt(0);
    __syncthreads();
    if (wid == 0 && kb + 32 < KD) {  // issue next tile into the other buffer
      tdm_load_2d(lds_off(&ldsA[buf ^ 1][0]), Xb + kb + 32, 32, 256, KD);
      tdm_load_2d(lds_off(&ldsW[buf ^ 1][0]), Wb + kb + 32, 32, 64, KD);
    }
    const _Float16* At = ldsA[buf];
    const _Float16* Wt = ldsW[buf];
    v16h a0 = load_afrag(At + (wid * 32 + l16) * 32, kbA);
    v16h a1 = load_afrag(At + (wid * 32 + 16 + l16) * 32, kbA);
#pragma unroll
    for (int ni = 0; ni < 4; ++ni) {
      v16h b = *(const v16h*)(Wt + (ni * 16 + l16) * 32 + kbB);
      c[0][ni] = wmma_f16(a0, b, c[0][ni]);
      c[1][ni] = wmma_f16(a1, b, c[1][ni]);
    }
  }

#pragma unroll
  for (int mi = 0; mi < 2; ++mi) {
#pragma unroll
    for (int ni = 0; ni < 4; ++ni) {
      const int col = col0 + ni * 16 + l16;
      const float bv = bias[col];
#pragma unroll
      for (int j = 0; j < 8; ++j) {
        const int row = row0 + mi * 16 + j + 8 * hseg;
        const float v = c[mi][ni][j] + bv;
        if (MODE == 2) {
          ((float*)dst)[(size_t)row * Hc + col] = v;
        } else {
          const int bb = row / Sc, ss = row % Sc;
          const int hh = col / HDc, dd = col % HDc;
          _Float16 hv = (_Float16)v;
          if (MODE == 0)
            ((_Float16*)dst)[(((size_t)(bb * NHc + hh)) * Sc + ss) * HDc + dd] = hv;
          else
            ((_Float16*)dst)[(((size_t)(bb * NHc + hh)) * HDc + dd) * Sc + ss] = hv;
        }
      }
    }
  }
}

// ---------------------------------------------------------------------------
// Flash attention: block = 4 waves = 4 q-tiles of one (b,h); all waves share
// TDM-staged, double-buffered 32-key K/V tiles in LDS. Online softmax with
// bias + key-padding mask; P re-laid-out through LDS for the P*V WMMA.
// Q,K: f16 [b,h,s,d]; Vt: f16 [b,h,d,s]; ctx out: f16 [b,s,h].
// ---------------------------------------------------------------------------
__global__ __launch_bounds__(128)
void attn_wmma(const _Float16* __restrict__ Q, const _Float16* __restrict__ K,
               const _Float16* __restrict__ Vt,
               const unsigned char* __restrict__ kpm,
               const float* __restrict__ bias, _Float16* __restrict__ ctx) {
  __shared__ __align__(128) _Float16 ldsK[2][32 * 64];   // 8 KB
  __shared__ __align__(128) _Float16 ldsV[2][64 * 32];   // 8 KB
  __shared__ __align__(32)  _Float16 ptile[4][16 * 32];  // 4 KB

  const int lane = threadIdx.x & 31;
  const int wid  = threadIdx.x >> 5;
  const int hseg = lane >> 4;
  const int l16  = lane & 15;
  const int kbA  = hseg ? 16 : 0;
  const int kbB  = hseg ? 16 : 0;

  const int bh = blockIdx.y;           // 0 .. B*NH-1
  const int bb = bh / NHc;
  const int hh = bh % NHc;
  const int q0 = (blockIdx.x * 4 + wid) * 16;

  // Q fragments (d = 0..31 and 32..63), fixed for the whole key loop
  const _Float16* qrow = Q + ((size_t)bh * Sc + q0 + l16) * HDc;
  const v16h aq0 = load_afrag(qrow, kbA);
  const v16h aq1 = load_afrag(qrow + 32, kbA);

  const _Float16* kbs = K  + (size_t)bh * Sc * HDc;
  const _Float16* vbs = Vt + (size_t)bh * HDc * Sc;
  const size_t biasbase = ((size_t)hh * Sc + q0) * Sc;
  _Float16* myp = ptile[wid];

  v8f o[4];
#pragma unroll
  for (int dt = 0; dt < 4; ++dt) o[dt] = (v8f){};
  float mr[8], lr[8];
#pragma unroll
  for (int j = 0; j < 8; ++j) { mr[j] = -1e30f; lr[j] = 0.0f; }

  const float scale = 0.125f;  // HD^-0.5

  if (wid == 0) {  // prologue: stage key block 0
    tdm_load_2d(lds_off(&ldsK[0][0]), kbs, 64, 32, HDc);
    tdm_load_2d(lds_off(&ldsV[0][0]), vbs, 32, 64, Sc);
  }

  for (int kb = 0; kb < Sc; kb += 32) {
    const int buf = (kb >> 5) & 1;
    if (wid == 0) __builtin_amdgcn_s_wait_tensorcnt(0);
    __syncthreads();
    if (wid == 0 && kb + 32 < Sc) {
      tdm_load_2d(lds_off(&ldsK[buf ^ 1][0]), kbs + (size_t)(kb + 32) * HDc, 64, 32, HDc);
      tdm_load_2d(lds_off(&ldsV[buf ^ 1][0]), vbs + kb + 32, 32, 64, Sc);
    }
    if (kb + 32 < Sc)  // prefetch next block of the (dominant) bias stream
      __builtin_prefetch(bias + biasbase + (size_t)(lane & 15) * Sc + kb + 32, 0, 1);

    const _Float16* Kt  = ldsK[buf];
    const _Float16* Vti = ldsV[buf];

    float lg[2][8];
    bool  msk[2];
#pragma unroll
    for (int t = 0; t < 2; ++t) {
      const int nrel = t * 16 + l16;       // key position within block
      const int n    = kb + nrel;          // global key position
      v16h bk0 = *(const v16h*)(Kt + nrel * 64 + kbB);
      v16h bk1 = *(const v16h*)(Kt + nrel * 64 + 32 + kbB);
      v8f s = (v8f){};
      s = wmma_f16(aq0, bk0, s);
      s = wmma_f16(aq1, bk1, s);
      msk[t] = kpm[(size_t)bb * Sc + n] != 0;
      const float* bp = bias + biasbase + n;
#pragma unroll
      for (int j = 0; j < 8; ++j) {
        const int m = j + 8 * hseg;
        const float v = s[j] * scale + bp[(size_t)m * Sc];
        lg[t][j] = msk[t] ? -1e30f : v;
      }
    }
    // online softmax (rows live in (VGPR j, lane-half); cols across 16 lanes)
#pragma unroll
    for (int j = 0; j < 8; ++j) {
      float rm = redmax16(fmaxf(lg[0][j], lg[1][j]));
      const float mn    = fmaxf(mr[j], rm);
      const float alpha = __expf(mr[j] - mn);
      const float p0 = msk[0] ? 0.0f : __expf(lg[0][j] - mn);
      const float p1 = msk[1] ? 0.0f : __expf(lg[1][j] - mn);
      const float rs = redsum16(p0 + p1);
      lr[j] = lr[j] * alpha + rs;
      mr[j] = mn;
      o[0][j] *= alpha; o[1][j] *= alpha; o[2][j] *= alpha; o[3][j] *= alpha;
      const int m = j + 8 * hseg;
      myp[m * 32 + l16]      = (_Float16)p0;
      myp[m * 32 + 16 + l16] = (_Float16)p1;
    }
    asm volatile("s_wait_dscnt 0" ::: "memory");  // DS in-order within wave
    v16h ap = load_afrag(myp + l16 * 32, kbA);
#pragma unroll
    for (int dt = 0; dt < 4; ++dt) {
      v16h bv = *(const v16h*)(Vti + (dt * 16 + l16) * 32 + kbB);
      o[dt] = wmma_f16(ap, bv, o[dt]);
    }
  }

  // normalize and store ctx [b, s, h] as f16
#pragma unroll
  for (int j = 0; j < 8; ++j) {
    const float inv = 1.0f / fmaxf(lr[j], 1e-6f);
    const int m = q0 + j + 8 * hseg;
    const size_t ob = ((size_t)bb * Sc + m) * Hc + hh * HDc;
#pragma unroll
    for (int dt = 0; dt < 4; ++dt)
      ctx[ob + dt * 16 + l16] = (_Float16)(o[dt][j] * inv);
  }
}

// ---------------------------------------------------------------------------
// Host side
// ---------------------------------------------------------------------------
extern "C" void kernel_launch(void* const* d_in, const int* in_sizes, int n_in,
                              void* d_out, int out_size, void* d_ws, size_t ws_size,
                              hipStream_t stream) {
  const float* query = (const float*)d_in[0];
  const float* key_t = (const float*)d_in[1];
  const float* value = (const float*)d_in[2];
  const unsigned char* kpm = (const unsigned char*)d_in[3];  // jnp bool = 1 byte
  const float* bias = (const float*)d_in[4];
  const float* q_w = (const float*)d_in[5];
  const float* q_b = (const float*)d_in[6];
  const float* k_w = (const float*)d_in[7];
  const float* k_b = (const float*)d_in[8];
  const float* v_w = (const float*)d_in[9];
  const float* v_b = (const float*)d_in[10];
  const float* o_w = (const float*)d_in[11];
  const float* o_b = (const float*)d_in[12];

  const size_t HHn  = (size_t)Hc * Hc;        // 1M elems
  const size_t BSHn = (size_t)Bc * Sc * Hc;   // 4M elems

  _Float16* ws    = (_Float16*)d_ws;
  _Float16* qw16  = ws;
  _Float16* kw16  = qw16 + HHn;
  _Float16* vw16  = kw16 + HHn;
  _Float16* ow16  = vw16 + HHn;
  _Float16* xq16  = ow16 + HHn;
  _Float16* xk16  = xq16 + BSHn;
  _Float16* xv16  = xk16 + BSHn;
  _Float16* Q16   = xv16 + BSHn;
  _Float16* K16   = Q16  + BSHn;
  _Float16* V16T  = K16  + BSHn;
  _Float16* ctx16 = V16T + BSHn;

  auto cvt = [&](const float* s, _Float16* d, size_t n) {
    int blocks = (int)((n / 4 + 255) / 256);
    cvt_f32_f16<<<blocks, 256, 0, stream>>>(s, d, n);
  };
  cvt(query, xq16, BSHn);
  cvt(key_t, xk16, BSHn);
  cvt(value, xv16, BSHn);
  cvt(q_w, qw16, HHn);
  cvt(k_w, kw16, HHn);
  cvt(v_w, vw16, HHn);
  cvt(o_w, ow16, HHn);

  dim3 ggrid(Bc * Sc / 256, Hc / 64);  // (16, 16)
  gemm_wmma<0><<<ggrid, 256, 0, stream>>>(xq16, qw16, q_b, (void*)Q16);
  gemm_wmma<0><<<ggrid, 256, 0, stream>>>(xk16, kw16, k_b, (void*)K16);
  gemm_wmma<1><<<ggrid, 256, 0, stream>>>(xv16, vw16, v_b, (void*)V16T);

  dim3 agrid(Sc / 16 / 4, Bc * NHc);   // (32, 32)
  attn_wmma<<<agrid, 128, 0, stream>>>(Q16, K16, V16T, kpm, bias, ctx16);

  gemm_wmma<2><<<ggrid, 256, 0, stream>>>(ctx16, ow16, o_b, d_out);
}
